// EpiGNN_6906307411999
// MI455X (gfx1250) — matrix operand order repaired and checked by reading, not compiled, not run on hardware
//
#include <hip/hip_runtime.h>
#include <hip/hip_bf16.h>
#include <math.h>

// ---------------------------------------------------------------------------
// EpiGNN forward, MI455X (gfx1250, wave32).
//   - attention stage reduced algebraically to per-row scalars (no BxMxM attn)
//   - la = n1 n2^T - n2 n1^T via v_wmma_f32_16x16x32_f16 -> binary mask
//   - lap (f16) @ Z via v_wmma_f32_16x16x32_f16, K=512 in 16 chunks,
//     Z stored transposed (B,48,512) so B-fragments are contiguous 128b loads
// ---------------------------------------------------------------------------

#define Bn 64
#define Tt 14
#define Mm 512
#define NFf 5
#define Kc 8
#define HIDR 40
#define HIDA 32
#define NLAYER 4
#define TOUT 7

typedef _Float16 h16;
typedef __attribute__((ext_vector_type(16))) _Float16 v16h;
typedef __attribute__((ext_vector_type(8)))  float    v8f;

__device__ __forceinline__ v8f wmma_f16(v16h a, v16h b, v8f c) {
  return __builtin_amdgcn_wmma_f32_16x16x32_f16(
      false, a, false, b, (short)0, c, false, false);
}

// A fragment (16x32, f16): rows across lanes (m = lane&15).
// lanes 0-15: K=0..15 (e -> K=e); lanes 16-31: K=16..31 (matches fragB)
__device__ __forceinline__ v16h fragA_rows(const h16* __restrict__ base, int ld,
                                           int kb, int lane) {
  const h16* p = base + (size_t)(lane & 15) * ld + kb + ((lane >> 4) << 4);
  v16h r;
#pragma unroll
  for (int e = 0; e < 16; ++e) r[e] = p[e];
  return r;
}

// B fragment (32x16, f16) where the source holds the *transposed* matrix
// row-major (rows = n, cols = k): B[k][n] = src[n*ld + k].
// lanes 0-15: K=0..15 (e -> K=e); lanes 16-31: K=16..31.
__device__ __forceinline__ v16h fragB_fromNK(const h16* __restrict__ base, int ld,
                                             int kb, int lane) {
  const h16* p = base + (size_t)(lane & 15) * ld + kb + ((lane >> 4) << 4);
  v16h r;
#pragma unroll
  for (int e = 0; e < 16; ++e) r[e] = p[e];
  return r;
}

// ---------------------------------------------------------------------------
// K1: degree of geographic adjacency
__global__ void k_deg(const float* __restrict__ adj, float* __restrict__ deg) {
  int m = blockIdx.x * blockDim.x + threadIdx.x;
  if (m < Mm) {
    float s = 0.f;
    for (int n = 0; n < Mm; ++n) s += adj[(size_t)m * Mm + n];
    deg[m] = s;
  }
}

// ---------------------------------------------------------------------------
// K2: RegionAwareConv backbone -> temp (B, M, 40), double tanh per reference
template <int KS, int DIL, int TO>
__device__ __forceinline__ void branch_pool(const float xv[NFf][Tt],
                                            const float* __restrict__ w,
                                            const float* __restrict__ bb,
                                            const float* __restrict__ gg,
                                            const float* __restrict__ be,
                                            float bni, float* __restrict__ out) {
  for (int c = 0; c < Kc; ++c) {
    float mx = -3.4e38f;
#pragma unroll
    for (int to = 0; to < TO; ++to) {
      float s = bb[c];
#pragma unroll
      for (int f = 0; f < NFf; ++f)
#pragma unroll
        for (int j = 0; j < KS; ++j)
          s += w[(c * NFf + f) * KS + j] * xv[f][to + j * DIL];
      float y = gg[c] * s * bni + be[c];
      mx = fmaxf(mx, y);
    }
    out[c] = tanhf(tanhf(mx));  // branch tanh, then tanh on concat
  }
}

__global__ void k_conv(const float* __restrict__ X,
                       const float* __restrict__ wl1, const float* __restrict__ bl1,
                       const float* __restrict__ gl1, const float* __restrict__ el1,
                       const float* __restrict__ wl2, const float* __restrict__ bl2,
                       const float* __restrict__ gl2, const float* __restrict__ el2,
                       const float* __restrict__ wp1, const float* __restrict__ bp1,
                       const float* __restrict__ gp1, const float* __restrict__ ep1,
                       const float* __restrict__ wp2, const float* __restrict__ bp2,
                       const float* __restrict__ gp2, const float* __restrict__ ep2,
                       const float* __restrict__ wg,  const float* __restrict__ bg,
                       const float* __restrict__ gg,  const float* __restrict__ eg,
                       float* __restrict__ temp) {
  int idx = blockIdx.x * blockDim.x + threadIdx.x;
  if (idx >= Bn * Mm) return;
  int b = idx >> 9, m = idx & (Mm - 1);
  float xv[NFf][Tt];
#pragma unroll
  for (int t = 0; t < Tt; ++t)
#pragma unroll
    for (int f = 0; f < NFf; ++f)
      xv[f][t] = X[(((size_t)b * Tt + t) * Mm + m) * NFf + f];
  const float bni = rsqrtf(1.0f + 1e-5f);
  float* out = temp + (size_t)idx * HIDR;
  branch_pool<3, 1, 12>(xv, wl1, bl1, gl1, el1, bni, out + 0);
  branch_pool<5, 1, 10>(xv, wl2, bl2, gl2, el2, bni, out + 8);
  branch_pool<3, 2, 10>(xv, wp1, bp1, gp1, ep1, bni, out + 16);
  branch_pool<5, 2, 6>(xv, wp2, bp2, gp2, ep2, bni, out + 24);
  branch_pool<14, 1, 1>(xv, wg, bg, gg, eg, bni, out + 32);
}

// ---------------------------------------------------------------------------
// K3a: q, k projections (B, M, 32)
__global__ void k_qk(const float* __restrict__ temp,
                     const float* __restrict__ wqw, const float* __restrict__ wqb,
                     const float* __restrict__ wkw, const float* __restrict__ wkb,
                     float* __restrict__ q, float* __restrict__ k) {
  int idx = blockIdx.x * blockDim.x + threadIdx.x;
  if (idx >= Bn * Mm) return;
  float tr[HIDR];
#pragma unroll
  for (int h = 0; h < HIDR; ++h) tr[h] = temp[(size_t)idx * HIDR + h];
  for (int a = 0; a < HIDA; ++a) {
    float sq = wqb[a], sk = wkb[a];
#pragma unroll
    for (int h = 0; h < HIDR; ++h) {
      sq += tr[h] * wqw[h * HIDA + a];
      sk += tr[h] * wkw[h * HIDA + a];
    }
    q[(size_t)idx * HIDA + a] = sq;
    k[(size_t)idx * HIDA + a] = sk;
  }
}

// K3b: per-batch  ksum = sum_m k[m,:]  and Gram G = k^T k (32x32)
__global__ void k_gram(const float* __restrict__ k, float* __restrict__ ksum,
                       float* __restrict__ G) {
  int b = blockIdx.x, t = threadIdx.x;
  __shared__ float kk[HIDA];
  int a0 = (t * 4) >> 5, c0 = (t * 4) & 31;  // this thread owns G[a0][c0..c0+3]
  float g0 = 0.f, g1 = 0.f, g2 = 0.f, g3 = 0.f, ks = 0.f;
  for (int m = 0; m < Mm; ++m) {
    if (t < HIDA) kk[t] = k[((size_t)b * Mm + m) * HIDA + t];
    __syncthreads();
    float ka = kk[a0];
    g0 += ka * kk[c0];  g1 += ka * kk[c0 + 1];
    g2 += ka * kk[c0 + 2]; g3 += ka * kk[c0 + 3];
    if (t < HIDA) ks += kk[t];
    __syncthreads();
  }
  float* Gb = G + (size_t)b * HIDA * HIDA;
  Gb[t * 4 + 0] = g0; Gb[t * 4 + 1] = g1; Gb[t * 4 + 2] = g2; Gb[t * 4 + 3] = g3;
  if (t < HIDA) ksum[(size_t)b * HIDA + t] = ks;
}

// K3c: feat = temp + r*t_w + t_b + deg*s_w + s_b, with
//   r = (q . ksum) / max(sqrt(q G q^T), 1e-12)   (== sum of L2-normalized attn row)
__global__ void k_feat(const float* __restrict__ temp, const float* __restrict__ q,
                       const float* __restrict__ ksum, const float* __restrict__ G,
                       const float* __restrict__ deg,
                       const float* __restrict__ tw, const float* __restrict__ tb,
                       const float* __restrict__ sw, const float* __restrict__ sb,
                       float* __restrict__ feat) {
  int idx = blockIdx.x * blockDim.x + threadIdx.x;
  if (idx >= Bn * Mm) return;
  int b = idx >> 9, m = idx & (Mm - 1);
  float qr[HIDA];
#pragma unroll
  for (int a = 0; a < HIDA; ++a) qr[a] = q[(size_t)idx * HIDA + a];
  const float* Gb = G + (size_t)b * HIDA * HIDA;
  const float* kb = ksum + (size_t)b * HIDA;
  float s1 = 0.f, s2 = 0.f;
  for (int a = 0; a < HIDA; ++a) {
    float qa = q[(size_t)idx * HIDA + a];
    s1 += qa * kb[a];
    float gq = 0.f;
#pragma unroll
    for (int a2 = 0; a2 < HIDA; ++a2) gq += Gb[a * HIDA + a2] * qr[a2];
    s2 += qa * gq;
  }
  float r = s1 / fmaxf(sqrtf(fmaxf(s2, 0.f)), 1e-12f);
  float dm = deg[m];
  for (int c = 0; c < HIDR; ++c)
    feat[(size_t)idx * HIDR + c] =
        temp[(size_t)idx * HIDR + c] + r * tw[c] + tb[c] + dm * sw[c] + sb[c];
}

// ---------------------------------------------------------------------------
// K4: n1 = tanh(temp@gl1+b1), n2 = tanh(temp@gl2+b2), f16, K padded 40->64
__global__ void k_n12(const float* __restrict__ temp,
                      const float* __restrict__ w1, const float* __restrict__ b1,
                      const float* __restrict__ w2, const float* __restrict__ b2,
                      h16* __restrict__ n1h, h16* __restrict__ n2h) {
  int idx = blockIdx.x * blockDim.x + threadIdx.x;
  if (idx >= Bn * Mm) return;
  float tr[HIDR];
#pragma unroll
  for (int h = 0; h < HIDR; ++h) tr[h] = temp[(size_t)idx * HIDR + h];
  h16* o1 = n1h + (size_t)idx * 64;
  h16* o2 = n2h + (size_t)idx * 64;
  for (int c = 0; c < HIDR; ++c) {
    float s1 = b1[c], s2 = b2[c];
#pragma unroll
    for (int h = 0; h < HIDR; ++h) {
      s1 += tr[h] * w1[h * HIDR + c];
      s2 += tr[h] * w2[h * HIDR + c];
    }
    o1[c] = (h16)tanhf(s1);
    o2[c] = (h16)tanhf(s2);
  }
  for (int c = HIDR; c < 64; ++c) { o1[c] = (h16)0.f; o2[c] = (h16)0.f; }
}

// K5: la tile = n1 n2^T - n2 n1^T  (WMMA f16, K=64 padded);
//     mask = (la>0) | (adj>0), branchless (comb>0 exact equivalence)
__global__ void k_la_mask(const h16* __restrict__ n1h, const h16* __restrict__ n2h,
                          const float* __restrict__ adj, unsigned char* __restrict__ mask) {
  int lane = threadIdx.x & 31;
  int ntile = blockIdx.x * 4 + (threadIdx.x >> 5);
  int mtile = blockIdx.y, b = blockIdx.z;
  const h16* A1 = n1h + ((size_t)b * Mm + mtile * 16) * 64;
  const h16* A2 = n2h + ((size_t)b * Mm + mtile * 16) * 64;
  const h16* B1 = n1h + ((size_t)b * Mm + ntile * 16) * 64;
  const h16* B2 = n2h + ((size_t)b * Mm + ntile * 16) * 64;
  v8f s = {}, t = {};
#pragma unroll
  for (int kc = 0; kc < 64; kc += 32) {
    s = wmma_f16(fragA_rows(A1, 64, kc, lane), fragB_fromNK(B2, 64, kc, lane), s);
    t = wmma_f16(fragA_rows(A2, 64, kc, lane), fragB_fromNK(B1, 64, kc, lane), t);
  }
  int ng = ntile * 16 + (lane & 15);
  int m0 = mtile * 16 + ((lane >> 4) << 3);
  // Unconditional adj loads + bitwise OR: no EXEC divergence around stores.
  float av[8];
#pragma unroll
  for (int r = 0; r < 8; ++r) av[r] = adj[(size_t)(m0 + r) * Mm + ng];
#pragma unroll
  for (int r = 0; r < 8; ++r) {
    float la = s[r] - t[r];
    int mk = (int)(la > 0.f) | (int)(av[r] > 0.f);
    mask[((size_t)b * Mm + (m0 + r)) * Mm + ng] = (unsigned char)mk;
  }
}

// K6: lap row = mask / rowcount, stored f16
__global__ void k_lap(const unsigned char* __restrict__ mask, h16* __restrict__ laph) {
  size_t row = (size_t)blockIdx.y * Mm + blockIdx.x;  // b*M + m
  const unsigned char* mr = mask + row * Mm;
  int t = threadIdx.x;
  __shared__ int sc[256];
  int c0 = mr[t], c1 = mr[t + 256];
  sc[t] = c0 + c1;
  __syncthreads();
  for (int off = 128; off > 0; off >>= 1) {
    if (t < off) sc[t] += sc[t + off];
    __syncthreads();
  }
  float dinv = 1.0f / ((float)sc[0] + 1e-12f);
  h16 dv = (h16)dinv;
  h16* lr = laph + row * Mm;
  lr[t] = c0 ? dv : (h16)0.f;
  lr[t + 256] = c1 ? dv : (h16)0.f;
}

// K7a: Zt = (src @ gcn_w[i])^T stored as (B, 48, 512) f16, rows 40..47 zero.
// Transposed store so K7b B-fragments are contiguous 128-bit loads.
__global__ void k_gcnZ(const float* __restrict__ src, const float* __restrict__ W,
                       h16* __restrict__ Zt) {
  int idx = blockIdx.x * blockDim.x + threadIdx.x;
  if (idx >= Bn * Mm) return;
  int b = idx >> 9, m = idx & (Mm - 1);
  float tr[HIDR];
#pragma unroll
  for (int h = 0; h < HIDR; ++h) tr[h] = src[(size_t)idx * HIDR + h];
  h16* z = Zt + (size_t)b * 48 * Mm + m;
  for (int c = 0; c < HIDR; ++c) {
    float s = 0.f;
#pragma unroll
    for (int h = 0; h < HIDR; ++h) s += tr[h] * W[h * HIDR + c];
    z[(size_t)c * Mm] = (h16)s;
  }
  for (int c = HIDR; c < 48; ++c) z[(size_t)c * Mm] = (h16)0.f;
}

// K7b: ns = elu(lap @ Z + b)   (WMMA f16, K=512 in 16 chunks, Zt is N-major)
__global__ void k_gcnY(const h16* __restrict__ laph, const h16* __restrict__ Zt,
                       const float* __restrict__ bias, float* __restrict__ ns) {
  int lane = threadIdx.x & 31;
  int ht = blockIdx.x, mt = blockIdx.y, b = blockIdx.z;
  const h16* Arow = laph + ((size_t)b * Mm + mt * 16) * Mm;
  const h16* Brow = Zt + ((size_t)b * 48 + ht * 16) * Mm;  // rows = h, cols = k
  v8f acc = {};
  for (int kc = 0; kc < Mm; kc += 32)
    acc = wmma_f16(fragA_rows(Arow, Mm, kc, lane),
                   fragB_fromNK(Brow, Mm, kc, lane), acc);
  int h = ht * 16 + (lane & 15);
  if (h < HIDR) {
    float bb = bias[h];
#pragma unroll
    for (int r = 0; r < 8; ++r) {
      int mg = mt * 16 + r + ((lane >> 4) << 3);
      float y = acc[r] + bb;
      ns[((size_t)b * Mm + mg) * HIDR + h] = (y > 0.f) ? y : expm1f(y);
    }
  }
}

// K8: out = concat(ns, feat) @ out_w + out_b
__global__ void k_out(const float* __restrict__ ns, const float* __restrict__ feat,
                      const float* __restrict__ ow, const float* __restrict__ ob,
                      float* __restrict__ out) {
  int idx = blockIdx.x * blockDim.x + threadIdx.x;
  if (idx >= Bn * Mm) return;
  float acc[TOUT];
#pragma unroll
  for (int j = 0; j < TOUT; ++j) acc[j] = ob[j];
  for (int h = 0; h < HIDR; ++h) {
    float v = ns[(size_t)idx * HIDR + h];
#pragma unroll
    for (int j = 0; j < TOUT; ++j) acc[j] += v * ow[h * TOUT + j];
  }
  for (int h = 0; h < HIDR; ++h) {
    float v = feat[(size_t)idx * HIDR + h];
#pragma unroll
    for (int j = 0; j < TOUT; ++j) acc[j] += v * ow[(HIDR + h) * TOUT + j];
  }
#pragma unroll
  for (int j = 0; j < TOUT; ++j) out[(size_t)idx * TOUT + j] = acc[j];
}

// ---------------------------------------------------------------------------
// Workspace layout (bytes). Lifetime-overlapped regions; total ~66.3 MB.
static const size_t OF_TEMP = 0;                         // f32 B*M*40   (5242880)
static const size_t OF_FEAT = OF_TEMP + 5242880;         // f32 B*M*40
static const size_t OF_LAPH = OF_FEAT + 5242880;         // f16 B*M*M   (33554432)
static const size_t OF_DEG  = OF_LAPH + 33554432;        // f32 512 (padded)
static const size_t OF_KSUM = OF_DEG + 4096;             // f32 64*32
static const size_t OF_G    = OF_KSUM + 8192;            // f32 64*32*32
static const size_t OF_RA   = OF_G + 262144;             // q,k (f32) -> n1h,n2h (f16)
static const size_t OF_RB   = OF_RA + 8388608;           // mask (u8) -> Zt (f16) + ns (f32)

extern "C" void kernel_launch(void* const* d_in, const int* in_sizes, int n_in,
                              void* d_out, int out_size, void* d_ws, size_t ws_size,
                              hipStream_t stream) {
  (void)in_sizes; (void)n_in; (void)out_size; (void)ws_size;
  const float* X    = (const float*)d_in[0];
  const float* adj  = (const float*)d_in[1];
  const float* w_l1 = (const float*)d_in[2];  const float* b_l1 = (const float*)d_in[3];
  const float* g_l1 = (const float*)d_in[4];  const float* e_l1 = (const float*)d_in[5];
  const float* w_l2 = (const float*)d_in[6];  const float* b_l2 = (const float*)d_in[7];
  const float* g_l2 = (const float*)d_in[8];  const float* e_l2 = (const float*)d_in[9];
  const float* w_p1 = (const float*)d_in[10]; const float* b_p1 = (const float*)d_in[11];
  const float* g_p1 = (const float*)d_in[12]; const float* e_p1 = (const float*)d_in[13];
  const float* w_p2 = (const float*)d_in[14]; const float* b_p2 = (const float*)d_in[15];
  const float* g_p2 = (const float*)d_in[16]; const float* e_p2 = (const float*)d_in[17];
  const float* w_g  = (const float*)d_in[18]; const float* b_g  = (const float*)d_in[19];
  const float* g_g  = (const float*)d_in[20]; const float* e_g  = (const float*)d_in[21];
  const float* wq_w = (const float*)d_in[22]; const float* wq_b = (const float*)d_in[23];
  const float* wk_w = (const float*)d_in[24]; const float* wk_b = (const float*)d_in[25];
  const float* t_w  = (const float*)d_in[26]; const float* t_b  = (const float*)d_in[27];
  const float* s_w  = (const float*)d_in[28]; const float* s_b  = (const float*)d_in[29];
  /* d_in[30] = d_gate: provably unused (only feeds comb>0, sigmoid>0 always) */
  const float* gl1_w = (const float*)d_in[31]; const float* gl1_b = (const float*)d_in[32];
  const float* gl2_w = (const float*)d_in[33]; const float* gl2_b = (const float*)d_in[34];
  const float* gcn_w = (const float*)d_in[35]; const float* gcn_b = (const float*)d_in[36];
  const float* out_w = (const float*)d_in[37]; const float* out_b = (const float*)d_in[38];

  char* ws = (char*)d_ws;
  float* temp = (float*)(ws + OF_TEMP);
  float* feat = (float*)(ws + OF_FEAT);
  h16*   laph = (h16*)(ws + OF_LAPH);
  float* deg  = (float*)(ws + OF_DEG);
  float* ksum = (float*)(ws + OF_KSUM);
  float* G    = (float*)(ws + OF_G);
  float* q    = (float*)(ws + OF_RA);                 // dead after k_feat
  float* k    = (float*)(ws + OF_RA + 4194304);
  h16*   n1h  = (h16*)(ws + OF_RA);                   // reuses q/k region
  h16*   n2h  = (h16*)(ws + OF_RA + 4194304);
  unsigned char* mask = (unsigned char*)(ws + OF_RB); // dead after k_lap
  h16*   Zt   = (h16*)(ws + OF_RB);                   // reuses mask region
  float* ns   = (float*)(ws + OF_RB + 4194304);
  float* outp = (float*)d_out;

  dim3 blk(256);
  int gBM = (Bn * Mm) / 256;  // 128 blocks

  k_deg<<<dim3(2), blk, 0, stream>>>(adj, deg);
  k_conv<<<dim3(gBM), blk, 0, stream>>>(X,
      w_l1, b_l1, g_l1, e_l1, w_l2, b_l2, g_l2, e_l2,
      w_p1, b_p1, g_p1, e_p1, w_p2, b_p2, g_p2, e_p2,
      w_g, b_g, g_g, e_g, temp);
  k_qk<<<dim3(gBM), blk, 0, stream>>>(temp, wq_w, wq_b, wk_w, wk_b, q, k);
  k_gram<<<dim3(Bn), blk, 0, stream>>>(k, ksum, G);
  k_feat<<<dim3(gBM), blk, 0, stream>>>(temp, q, ksum, G, deg, t_w, t_b, s_w, s_b, feat);
  k_n12<<<dim3(gBM), blk, 0, stream>>>(temp, gl1_w, gl1_b, gl2_w, gl2_b, n1h, n2h);
  k_la_mask<<<dim3(8, 32, Bn), dim3(128), 0, stream>>>(n1h, n2h, adj, mask);
  k_lap<<<dim3(Mm, Bn), blk, 0, stream>>>(mask, laph);
  for (int i = 0; i < NLAYER; ++i) {
    const float* src = (i == 0) ? feat : ns;
    k_gcnZ<<<dim3(gBM), blk, 0, stream>>>(src, gcn_w + (size_t)i * HIDR * HIDR, Zt);
    k_gcnY<<<dim3(3, 32, Bn), dim3(32), 0, stream>>>(laph, Zt, gcn_b + (size_t)i * HIDR, ns);
  }
  k_out<<<dim3(gBM), blk, 0, stream>>>(ns, feat, out_w, out_b, outp);
}
